// GTG_Module_49555332661584
// MI455X (gfx1250) — matrix operand order repaired and checked by reading, not compiled, not run on hardware
//
#include <hip/hip_runtime.h>
#include <hip/hip_bf16.h>
#include <math.h>

// ---------------- problem constants ----------------
#define B_N    8192
#define D_DIM  512
#define C_N    100
#define NPAD   112          // 100 padded to 7 * 16 WMMA column tiles
#define N_LABL 4096
#define MAXIT  30
#define TOLV   1e-3f
#define CHUNK_K 256         // K-chunk staged into LDS by the Tensor Data Mover

// ---------------- WMMA / TDM types ----------------
typedef __attribute__((ext_vector_type(16))) __bf16       v16bf;
typedef __attribute__((ext_vector_type(8)))  float        v8f;
typedef __attribute__((ext_vector_type(4)))  unsigned int v4u;
typedef __attribute__((ext_vector_type(4)))  unsigned int u32x4;
typedef __attribute__((ext_vector_type(8)))  int          i32x8;
typedef __attribute__((ext_vector_type(4)))  int          i32x4;

union Frag { v4u u[2]; v16bf h; };

#define WMMA_BF16(a,b,c) \
  __builtin_amdgcn_wmma_f32_16x16x32_bf16(false,(a),false,(b),(short)0,(c),false,false)

__device__ __forceinline__ unsigned short f2bf(float x){
  unsigned int u = __float_as_uint(x);
  u = (u + 0x7FFFu + ((u >> 16) & 1u)) >> 16;   // round-to-nearest-even
  return (unsigned short)u;
}
__device__ __forceinline__ float bf2f(unsigned short h){
  return __uint_as_float(((unsigned int)h) << 16);
}

// A-operand fragment: source row-major [.., ld], lane&15 = row, two 16B chunks
// (contiguous K order matching B: lanes 0-15 hold K=0..15, lanes 16-31 hold K=16..31)
__device__ __forceinline__ v16bf load_frag_rows(const unsigned short* base, int ld,
                                                int row0, int k0){
  int lane = threadIdx.x & 31;
  int r = lane & 15, hi = lane >> 4;
  const unsigned short* p = base + (size_t)(row0 + r) * ld + (k0 + 16 * hi);
  Frag f;
  f.u[0] = *(const v4u*)(p);        // K = k0+16hi .. +7
  f.u[1] = *(const v4u*)(p + 8);    // K = k0+16hi+8 .. +7
  return f.h;
}
// B-operand fragment from K-contiguous ("transposed") storage baseT[col][k]:
// lane&15 = output column, 16 contiguous K values starting at k0+16*(lane>>4)
__device__ __forceinline__ v16bf load_frag_cols(const unsigned short* baseT, int ld,
                                                int col0, int k0){
  int lane = threadIdx.x & 31;
  int c = lane & 15, hi = lane >> 4;
  const unsigned short* p = baseT + (size_t)(col0 + c) * ld + (k0 + 16 * hi);
  Frag f;
  f.u[0] = *(const v4u*)(p);
  f.u[1] = *(const v4u*)(p + 8);
  return f.h;
}
// same, but from an LDS-staged tile Bs[NPAD][CHUNK_K]
__device__ __forceinline__ v16bf load_frag_cols_lds(const unsigned short* Bs,
                                                    int col0, int krel){
  int lane = threadIdx.x & 31;
  int c = lane & 15, hi = lane >> 4;
  const unsigned short* p = Bs + (col0 + c) * CHUNK_K + krel + 16 * hi;
  Frag f;
  f.u[0] = *(const v4u*)(p);
  f.u[1] = *(const v4u*)(p + 8);
  return f.h;
}

// ---------------- Tensor Data Mover: 2-D tile load, global -> LDS ----------------
// D# per CDNA5 ISA: group0 = {count/lds_addr/global_addr/type},
// group1 = {data_size, tensor_dim0/1, tile_dim0/1, dim0_stride}. Groups 2/3 zero (2-D).
#if __has_builtin(__builtin_amdgcn_tensor_load_to_lds)
#define HAVE_TDM 1
__device__ __forceinline__ void tdm_load_2d(unsigned lds_addr, const void* gaddr,
                                            unsigned tensor_d0, unsigned tensor_d1,
                                            unsigned tile_d0, unsigned tile_d1,
                                            unsigned long long d0_stride){
  unsigned long long ga = (unsigned long long)(size_t)gaddr;
  u32x4 g0;
  g0[0] = 1u;                                              // count=1, user descriptor
  g0[1] = lds_addr;                                        // LDS byte address
  g0[2] = (unsigned)(ga & 0xFFFFFFFFu);                    // global_addr[31:0]
  g0[3] = (unsigned)((ga >> 32) & 0x01FFFFFFu) | (2u << 30); // addr[56:32] | type=2
  i32x8 g1;
  g1[0] = (int)(1u << 16);                                 // data_size=1 (2B), wg_mask=0
  g1[1] = (int)((tensor_d0 & 0xFFFFu) << 16);              // tensor_dim0 bits 79:48
  g1[2] = (int)((tensor_d0 >> 16) | ((tensor_d1 & 0xFFFFu) << 16)); // + tensor_dim1
  g1[3] = (int)((tensor_d1 >> 16) | (tile_d0 << 16));      // tile_dim0 bits 127:112
  g1[4] = (int)(tile_d1 & 0xFFFFu);                        // tile_dim1 ; tile_dim2=0
  g1[5] = (int)(unsigned)(d0_stride & 0xFFFFFFFFull);      // dim0_stride bits 191:160
  g1[6] = (int)(unsigned)((d0_stride >> 32) & 0xFFFFull);  // dim0_stride bits 207:192
  g1[7] = 0;
  i32x4 gz = {0, 0, 0, 0};
#if defined(__clang_major__) && (__clang_major__ >= 23)
  i32x8 gz8 = {0, 0, 0, 0, 0, 0, 0, 0};
  __builtin_amdgcn_tensor_load_to_lds(g0, g1, gz, gz, gz8, 0);
#else
  __builtin_amdgcn_tensor_load_to_lds(g0, g1, gz, gz, 0);
#endif
}
#else
#define HAVE_TDM 0
#endif

// ---------------- setup kernels ----------------
__global__ void k_init_scalars(float* sumA, float* meanA, float* errsq, int* flag){
  *sumA = 0.f; *meanA = 0.f; *errsq = 0.f; *flag = 0;
}

// row-normalize embedds -> E (bf16) ; also raw embedds -> Emb (bf16) for MLP
__global__ void k_rownorm(const float* __restrict__ emb,
                          unsigned short* __restrict__ E,
                          unsigned short* __restrict__ Emb){
  __shared__ float red[128];
  int r = blockIdx.x, t = threadIdx.x;
  float ss = 0.f;
  for (int c = t; c < D_DIM; c += 128){ float v = emb[(size_t)r*D_DIM + c]; ss += v*v; }
  red[t] = ss; __syncthreads();
  for (int s = 64; s > 0; s >>= 1){ if (t < s) red[t] += red[t+s]; __syncthreads(); }
  float inv = 1.f / fmaxf(sqrtf(red[0]), 1e-12f);
  for (int c = t; c < D_DIM; c += 128){
    float v = emb[(size_t)r*D_DIM + c];
    E  [(size_t)r*D_DIM + c] = f2bf(v * inv);
    Emb[(size_t)r*D_DIM + c] = f2bf(v);
  }
}

// W[K][N] f32 -> WT[N][K] bf16 (K-contiguous for B-fragments)
__global__ void k_transpose_w(const float* __restrict__ W, unsigned short* __restrict__ WT,
                              int K, int N){
  int idx = blockIdx.x * blockDim.x + threadIdx.x;
  if (idx >= K * N) return;
  int k = idx / N, n = idx % N;
  WT[(size_t)n * K + k] = f2bf(W[idx]);
}

// X0: labeled rows one-hot, unlabeled 1/C ; XbT bf16 transposed w/ zero padding ; y_acc = 0
__global__ void k_init_state(const int* __restrict__ labels, float* __restrict__ X,
                             unsigned short* __restrict__ XbT, float* __restrict__ yacc){
  int r = blockIdx.x * blockDim.x + threadIdx.x;
  if (r >= B_N) return;
  int lab = labels[r];
  bool isl = r < N_LABL;
  for (int c = 0; c < NPAD; ++c){
    float x0 = (c < C_N) ? (isl ? (lab == c ? 1.f : 0.f) : (1.f / (float)C_N)) : 0.f;
    if (c < C_N) X[(size_t)r * C_N + c] = x0;
    XbT[(size_t)c * B_N + r] = f2bf(x0);
  }
  yacc[r] = 0.f;
}

// ---------------- A = clamp(E E^T, diag->0) ; partial sums for mean ----------------
// block = 128 threads (4 waves); wave tile 16 rows x 64 cols; grid (128,128)
__global__ void k_syrk(const unsigned short* __restrict__ E, unsigned short* __restrict__ A,
                       float* __restrict__ sumA){
  __shared__ float red[128];
  int w  = threadIdx.x >> 5;
  int i0 = blockIdx.x * 64 + w * 16;
  int j0 = blockIdx.y * 64;
  v8f acc[4] = {};
  for (int k0 = 0; k0 < D_DIM; k0 += 32){
    v16bf a = load_frag_rows(E, D_DIM, i0, k0);
    #pragma unroll
    for (int t = 0; t < 4; ++t){
      v16bf b = load_frag_cols(E, D_DIM, j0 + 16 * t, k0);  // col j of B == row j of E
      acc[t] = WMMA_BF16(a, b, acc[t]);
    }
  }
  int lane = threadIdx.x & 31;
  int n = lane & 15, mh = (lane >> 4) * 8;   // C/D layout: VGPR v -> M = v + 8*(lane>>4)
  float ls = 0.f;
  #pragma unroll
  for (int t = 0; t < 4; ++t){
    int j = j0 + 16 * t + n;
    #pragma unroll
    for (int v = 0; v < 8; ++v){
      int i = i0 + mh + v;
      float a = acc[t][v];
      a = fminf(fmaxf(a, 0.f), 1.f);
      if (i == j) a = 0.f;                   // fill_diagonal_(0)
      ls += a;
      A[(size_t)i * B_N + j] = f2bf(a);
    }
  }
  red[threadIdx.x] = ls; __syncthreads();
  for (int s = 64; s > 0; s >>= 1){ if (threadIdx.x < s) red[threadIdx.x] += red[threadIdx.x + s]; __syncthreads(); }
  if (threadIdx.x == 0) atomicAdd(sumA, red[0]);
}

__global__ void k_mean(const float* sumA, float* meanA){
  *meanA = *sumA / ((float)B_N * (float)B_N);
}

// mean threshold + diversity: A = (a < mean) ? 1 : 1 - a   (diag 0 -> 1, as in reference)
__global__ void k_thresh(unsigned short* __restrict__ A, const float* __restrict__ meanA){
  size_t idx = (size_t)blockIdx.x * 256 + threadIdx.x;
  float m = *meanA;
  float a = bf2f(A[idx]);
  A[idx] = f2bf((a < m) ? 1.f : 1.f - a);
}

// ---------------- P = A @ Xb ----------------
// A (bf16, 128 MB, L2-resident) streamed from global; Xb K-chunks DMA'd into LDS by the
// Tensor Data Mover, double-buffered and overlapped with WMMA compute via TENSORcnt.
// block = 128 threads (4 waves); wave tile 32 rows x 112 cols (2x7 wmma accums); grid 64
__global__ void k_gemm_AX(const unsigned short* __restrict__ A,
                          const unsigned short* __restrict__ XbT,
                          float* __restrict__ P, const int* __restrict__ flag){
  if (*flag) return;                         // device-side convergence gate (graph-safe)
  __shared__ unsigned short Bs[2][NPAD * CHUNK_K];   // 2 x 57 KB
  int w    = threadIdx.x >> 5;
  int i0   = blockIdx.x * 128 + w * 32;
  int lane = threadIdx.x & 31;
  v8f acc[2][7] = {};

#if HAVE_TDM
  if (w == 0)                               // one wave issues the DMA (EXEC-independent)
    tdm_load_2d((unsigned)(size_t)(void*)&Bs[0][0], XbT,
                B_N, NPAD, CHUNK_K, NPAD, B_N);
#endif

  for (int kc = 0; kc < B_N; kc += CHUNK_K){
    int cur = (kc / CHUNK_K) & 1;
#if HAVE_TDM
    if (w == 0){
      if (kc + CHUNK_K < B_N){              // prefetch next chunk into the other buffer
        tdm_load_2d((unsigned)(size_t)(void*)&Bs[cur ^ 1][0], XbT + (kc + CHUNK_K),
                    B_N, NPAD, CHUNK_K, NPAD, B_N);
        __builtin_amdgcn_s_wait_tensorcnt(1);   // in-order: current chunk's DMA is done
      } else {
        __builtin_amdgcn_s_wait_tensorcnt(0);   // last chunk: drain
      }
    }
#else
    // fallback: cooperative staged copy (no TDM builtin on this toolchain)
    {
      const v4u* src = (const v4u*)(XbT);
      v4u* dst = (v4u*)&Bs[cur][0];
      for (int e = threadIdx.x; e < NPAD * CHUNK_K / 8; e += 128){
        int row = e / (CHUNK_K / 8), col = e % (CHUNK_K / 8);
        dst[e] = src[((size_t)row * B_N + kc) / 8 + col];
      }
    }
#endif
    __syncthreads();                        // DMA visible to all waves
    const unsigned short* B = &Bs[cur][0];
    for (int k0 = 0; k0 < CHUNK_K; k0 += 32){
      __builtin_prefetch(A + (size_t)(i0 + (lane & 15)) * B_N + kc + k0 + 512, 0, 1);
      v16bf a0 = load_frag_rows(A, B_N, i0,      kc + k0);
      v16bf a1 = load_frag_rows(A, B_N, i0 + 16, kc + k0);
      #pragma unroll
      for (int t = 0; t < 7; ++t){
        v16bf b = load_frag_cols_lds(B, 16 * t, k0);
        acc[0][t] = WMMA_BF16(a0, b, acc[0][t]);
        acc[1][t] = WMMA_BF16(a1, b, acc[1][t]);
      }
    }
    __syncthreads();                        // all waves done with Bs[cur] before rewrite
  }

  int n = lane & 15, mh = (lane >> 4) * 8;
  #pragma unroll
  for (int rt = 0; rt < 2; ++rt)
    #pragma unroll
    for (int t = 0; t < 7; ++t)
      #pragma unroll
      for (int v = 0; v < 8; ++v)
        P[(size_t)(i0 + 16 * rt + mh + v) * NPAD + 16 * t + n] = acc[rt][t][v];
}

// ---------------- replicator update + entropy + err ----------------
// block = 128 (4 waves), one row per wave; grid 2048
__global__ void k_update(const float* __restrict__ P, float* __restrict__ X,
                         unsigned short* __restrict__ XbT, float* __restrict__ yacc,
                         float* __restrict__ errsq, const int* __restrict__ flag){
  if (*flag) return;
  __shared__ float red[4];
  int w = threadIdx.x >> 5;
  int lane = threadIdx.x & 31;
  int r = blockIdx.x * 4 + w;
  float xn[4], xo[4], s = 0.f;
  #pragma unroll
  for (int q = 0; q < 4; ++q){
    int c = lane + 32 * q;
    float x = (c < C_N) ? X[(size_t)r * C_N + c] : 0.f;
    float p = (c < C_N) ? P[(size_t)r * NPAD + c] : 0.f;
    xo[q] = x;
    xn[q] = x * p;
    s += xn[q];
  }
  for (int o = 16; o > 0; o >>= 1) s += __shfl_down(s, o, 32);   // wave32 reduce
  s = __shfl(s, 0, 32);
  float inv = 1.f / s;
  float ent = 0.f, d2 = 0.f;
  #pragma unroll
  for (int q = 0; q < 4; ++q){
    int c = lane + 32 * q;
    if (c < C_N){
      float v = xn[q] * inv;
      ent -= v * log2f(v + 1e-20f);
      float dd = v - xo[q];
      d2 += dd * dd;
      X[(size_t)r * C_N + c] = v;
      XbT[(size_t)c * B_N + r] = f2bf(v);
    }
  }
  for (int o = 16; o > 0; o >>= 1){ ent += __shfl_down(ent, o, 32); d2 += __shfl_down(d2, o, 32); }
  if (lane == 0){
    if (r >= N_LABL) yacc[r] += ent;         // hist column for this iteration (labeled -> 0)
    red[w] = d2;
  }
  __syncthreads();
  if (threadIdx.x == 0) atomicAdd(errsq, red[0] + red[1] + red[2] + red[3]);
}

__global__ void k_check(float* errsq, int* flag){
  if (sqrtf(*errsq) <= TOLV) *flag = 1;
  *errsq = 0.f;
}

// ---------------- MLP: generic WMMA GEMM + bias (+relu), bf16 in/out ----------------
// block = 128 (4 waves); wave tile 16 rows x 64 cols; grid (M/64, N/64)
__global__ void k_gemm_bias_relu(const unsigned short* __restrict__ Abf,
                                 const unsigned short* __restrict__ WT,
                                 const float* __restrict__ bias,
                                 unsigned short* __restrict__ Cbf,
                                 int N, int K, int relu){
  int w  = threadIdx.x >> 5;
  int i0 = blockIdx.x * 64 + w * 16;
  int j0 = blockIdx.y * 64;
  v8f acc[4] = {};
  for (int k0 = 0; k0 < K; k0 += 32){
    v16bf a = load_frag_rows(Abf, K, i0, k0);
    #pragma unroll
    for (int t = 0; t < 4; ++t){
      v16bf b = load_frag_cols(WT, K, j0 + 16 * t, k0);
      acc[t] = WMMA_BF16(a, b, acc[t]);
    }
  }
  int lane = threadIdx.x & 31;
  int n = lane & 15, mh = (lane >> 4) * 8;
  #pragma unroll
  for (int t = 0; t < 4; ++t){
    int j = j0 + 16 * t + n;
    #pragma unroll
    for (int v = 0; v < 8; ++v){
      float h = acc[t][v] + bias[j];
      if (relu) h = fmaxf(h, 0.f);
      Cbf[(size_t)(i0 + mh + v) * N + j] = f2bf(h);
    }
  }
}

// last layer (N=1): y_pred[r] = H3[r,:] . Wc + bc
__global__ void k_final(const unsigned short* __restrict__ H3, const float* __restrict__ Wc,
                        const float* __restrict__ bc, float* __restrict__ out){
  int r = blockIdx.x * blockDim.x + threadIdx.x;
  if (r >= B_N) return;
  float s = bc[0];
  #pragma unroll 8
  for (int k = 0; k < 64; ++k) s += bf2f(H3[(size_t)r * 64 + k]) * Wc[k];
  out[r] = s;
}

__global__ void k_finalize(const float* __restrict__ yacc, float* __restrict__ out){
  int r = blockIdx.x * blockDim.x + threadIdx.x;
  if (r >= B_N) return;
  out[B_N + r]     = yacc[r] * (1.f / (float)MAXIT);   // mean over 30 hist columns
  out[2 * B_N + r] = (r < N_LABL) ? 1.f : 0.f;         // labeled_mask
}

// ---------------- launch ----------------
extern "C" void kernel_launch(void* const* d_in, const int* in_sizes, int n_in,
                              void* d_out, int out_size, void* d_ws, size_t ws_size,
                              hipStream_t stream) {
  const float* embedds = (const float*)d_in[1];
  const int*   labels  = (const int*)  d_in[2];
  const float* W1 = (const float*)d_in[3];  const float* b1 = (const float*)d_in[4];
  const float* W2 = (const float*)d_in[5];  const float* b2 = (const float*)d_in[6];
  const float* W3 = (const float*)d_in[7];  const float* b3 = (const float*)d_in[8];
  const float* Wc = (const float*)d_in[9];  const float* bc = (const float*)d_in[10];
  float* out = (float*)d_out;

  // workspace carve-out (256B aligned)
  size_t off = 0;
  char* base = (char*)d_ws;
  auto carve = [&](size_t bytes) -> char* {
    char* p = base + off; off += (bytes + 255) & ~(size_t)255; return p;
  };
  unsigned short* A    = (unsigned short*)carve((size_t)B_N * B_N * 2);   // 128 MB, L2-resident
  unsigned short* E    = (unsigned short*)carve((size_t)B_N * D_DIM * 2);
  unsigned short* Emb  = (unsigned short*)carve((size_t)B_N * D_DIM * 2);
  unsigned short* W1T  = (unsigned short*)carve((size_t)D_DIM * 256 * 2);
  unsigned short* W2T  = (unsigned short*)carve((size_t)256 * 128 * 2);
  unsigned short* W3T  = (unsigned short*)carve((size_t)128 * 64 * 2);
  unsigned short* H1   = (unsigned short*)carve((size_t)B_N * 256 * 2);
  unsigned short* H2   = (unsigned short*)carve((size_t)B_N * 128 * 2);
  unsigned short* H3   = (unsigned short*)carve((size_t)B_N * 64 * 2);
  unsigned short* XbT  = (unsigned short*)carve((size_t)NPAD * B_N * 2);
  float*          X    = (float*)carve((size_t)B_N * C_N * 4);
  float*          P    = (float*)carve((size_t)B_N * NPAD * 4);
  float*          yacc = (float*)carve((size_t)B_N * 4);
  float*          sumA  = (float*)carve(4);
  float*          meanA = (float*)carve(4);
  float*          errsq = (float*)carve(4);
  int*            flag  = (int*)  carve(4);

  k_init_scalars<<<1, 1, 0, stream>>>(sumA, meanA, errsq, flag);
  k_rownorm<<<B_N, 128, 0, stream>>>(embedds, E, Emb);
  k_transpose_w<<<(D_DIM * 256 + 255) / 256, 256, 0, stream>>>(W1, W1T, D_DIM, 256);
  k_transpose_w<<<(256 * 128 + 255) / 256, 256, 0, stream>>>(W2, W2T, 256, 128);
  k_transpose_w<<<(128 * 64 + 255) / 256, 256, 0, stream>>>(W3, W3T, 128, 64);
  k_init_state<<<(B_N + 255) / 256, 256, 0, stream>>>(labels, X, XbT, yacc);

  // A matrix: SYRK + global mean + threshold/diversity
  k_syrk<<<dim3(B_N / 64, B_N / 64), 128, 0, stream>>>(E, A, sumA);
  k_mean<<<1, 1, 0, stream>>>(sumA, meanA);
  k_thresh<<<(unsigned)((size_t)B_N * B_N / 256), 256, 0, stream>>>(A, meanA);

  // 30 replicator iterations, gated by device-side convergence flag
  for (int it = 0; it < MAXIT; ++it){
    k_gemm_AX<<<B_N / 128, 128, 0, stream>>>(A, XbT, P, flag);
    k_update <<<B_N / 4,   128, 0, stream>>>(P, X, XbT, yacc, errsq, flag);
    k_check  <<<1, 1, 0, stream>>>(errsq, flag);
  }

  // MLP head
  k_gemm_bias_relu<<<dim3(B_N / 64, 4), 128, 0, stream>>>(Emb, W1T, b1, H1, 256, D_DIM, 1);
  k_gemm_bias_relu<<<dim3(B_N / 64, 2), 128, 0, stream>>>(H1,  W2T, b2, H2, 128, 256, 1);
  k_gemm_bias_relu<<<dim3(B_N / 64, 1), 128, 0, stream>>>(H2,  W3T, b3, H3, 64, 128, 1);
  k_final   <<<(B_N + 255) / 256, 256, 0, stream>>>(H3, Wc, bc, out);
  k_finalize<<<(B_N + 255) / 256, 256, 0, stream>>>(yacc, out);
}
